// GroupedQueryAttentionWithRoPE_8297876816192
// MI455X (gfx1250) — matrix operand order repaired and checked by reading, not compiled, not run on hardware
//
#include <hip/hip_runtime.h>

// ---------------------------------------------------------------------------
// GroupedQueryAttentionWithRoPE for MI455X (gfx1250, wave32, WMMA)
// b=2, T=2048, D=2048, H=32, KVH=8, G=4, HD=64
// Compute-bound (~155 GFLOP vs ~150MB traffic @ 23.3TB/s) -> all matmuls via
// v_wmma_f32_16x16x32_bf16 (fp32 accumulate); tile staging via async
// global->LDS copies (ASYNCcnt) with double buffering where it pays.
// ---------------------------------------------------------------------------

typedef __attribute__((ext_vector_type(16))) __bf16 v16bf;
typedef __attribute__((ext_vector_type(8)))  float  v8f;
typedef __attribute__((ext_vector_type(4)))  int    v4i;

#define T_SEQ   2048
#define D_MODEL 2048
#define N_HEADS 32
#define KV_HEADS 8
#define HEAD_DIM 64
#define KV_D    (KV_HEADS * HEAD_DIM)   // 512
#define BATCH   2
#define M_ROWS  (BATCH * T_SEQ)         // 4096

// ---------------- async global->LDS helpers (CDNA5 path) -------------------
#if defined(__gfx1250__) && __has_builtin(__builtin_amdgcn_global_load_async_to_lds_b128)
#define GQA_HAVE_ASYNC_LDS 1
#else
#define GQA_HAVE_ASYNC_LDS 0
#endif

__device__ __forceinline__ void async_copy16(const __bf16* __restrict__ g,
                                             __bf16* __restrict__ l) {
#if GQA_HAVE_ASYNC_LDS
  // Signature (from clang diagnostic): (v4i AS1*, v4i AS3*, imm offset, imm cpol)
  __builtin_amdgcn_global_load_async_to_lds_b128(
      (__attribute__((address_space(1))) v4i*)g,
      (__attribute__((address_space(3))) v4i*)l, 0, 0);
#else
  *(uint4*)l = *(const uint4*)g;   // fallback through VGPRs
#endif
}

__device__ __forceinline__ void wait_async0() {
#if defined(__gfx1250__)
#if __has_builtin(__builtin_amdgcn_s_wait_asynccnt)
  __builtin_amdgcn_s_wait_asynccnt(0);
#else
  asm volatile("s_wait_asynccnt 0x0" ::: "memory");
#endif
#endif
}

// ------------------------------- cast f32 -> bf16 --------------------------
__global__ __launch_bounds__(256) void cast_f32_bf16(const float* __restrict__ s,
                                                     __bf16* __restrict__ d, int n) {
  int i = blockIdx.x * 256 + threadIdx.x;
  if (i < n) d[i] = (__bf16)s[i];
}

// ------------------------------- WMMA GEMM ---------------------------------
// C[M,N] = A[M,K] @ B[K,N], row-major, bf16 inputs, fp32 accumulate.
// Block: 256 threads = 8 waves as 4(M) x 2(N); block tile 128x128, K step 32.
// Double-buffered LDS, tiles staged with async global->LDS b128 copies.
template <bool F32OUT>
__global__ __launch_bounds__(256)
void gemm_bf16_wmma(const __bf16* __restrict__ A, const __bf16* __restrict__ B,
                    __bf16* __restrict__ Cb, float* __restrict__ Cf,
                    const float* __restrict__ bias, int M, int N, int K) {
  __shared__ __align__(16) __bf16 As[2][128 * 32];
  __shared__ __align__(16) __bf16 Bs[2][32 * 128];
  const int tid  = threadIdx.x;
  const int wave = tid >> 5, lane = tid & 31;
  const int half = lane >> 4, l16 = lane & 15;
  const int wm = wave >> 1, wn = wave & 1;
  const int bm = blockIdx.y * 128, bn = blockIdx.x * 128;

  // Issue async copies for one 128x32 A tile + 32x128 B tile into buffer `buf`.
  // 512 16-byte chunks per tile; 256 threads -> 2 chunks each per tile.
  auto issue_tile = [&](int k0, int buf) {
    #pragma unroll
    for (int c = 0; c < 2; ++c) {
      const int ca = tid + c * 256;               // A chunk id 0..511
      const int arow = ca >> 2, acol = (ca & 3) * 8;
      async_copy16(&A[(size_t)(bm + arow) * K + (k0 + acol)],
                   &As[buf][arow * 32 + acol]);
      const int brow = ca >> 4, bcol = (ca & 15) * 8;
      async_copy16(&B[(size_t)(k0 + brow) * N + (bn + bcol)],
                   &Bs[buf][brow * 128 + bcol]);
    }
  };

  v8f acc[2][4] = {};

  issue_tile(0, 0);
  wait_async0();
  __syncthreads();

  int buf = 0;
  for (int k0 = 0; k0 < K; k0 += 32) {
    if (k0 + 32 < K) issue_tile(k0 + 32, buf ^ 1);

    // B fragments: col N = lane%16, K(e) = half*16 + e
    v16bf bfr[4];
    #pragma unroll
    for (int ni = 0; ni < 4; ++ni) {
      const int n = wn * 64 + ni * 16 + l16;
      #pragma unroll
      for (int e = 0; e < 16; ++e)
        bfr[ni][e] = Bs[buf][(half * 16 + e) * 128 + n];
    }
    // A fragments: row M = lane%16, K(e) = half*16 + e
    #pragma unroll
    for (int mi = 0; mi < 2; ++mi) {
      v16bf a;
      const int m = wm * 32 + mi * 16 + l16;
      #pragma unroll
      for (int e = 0; e < 16; ++e) {
        const int kk = half * 16 + e;
        a[e] = As[buf][m * 32 + kk];
      }
      #pragma unroll
      for (int ni = 0; ni < 4; ++ni)
        acc[mi][ni] = __builtin_amdgcn_wmma_f32_16x16x32_bf16(
            false, a, false, bfr[ni], (short)0, acc[mi][ni], false, false);
    }

    wait_async0();      // next tile resident in LDS
    __syncthreads();    // also: everyone done reading `buf` before it is reused
    buf ^= 1;
  }

  // C/D layout: row M = r + 8*half, col N = lane%16
  #pragma unroll
  for (int mi = 0; mi < 2; ++mi)
    #pragma unroll
    for (int ni = 0; ni < 4; ++ni)
      #pragma unroll
      for (int r = 0; r < 8; ++r) {
        const int m = bm + wm * 32 + mi * 16 + r + 8 * half;
        const int n = bn + wn * 64 + ni * 16 + l16;
        const float v = acc[mi][ni][r];
        if (F32OUT) Cf[(size_t)m * N + n] = v + (bias ? bias[n] : 0.0f);
        else        Cb[(size_t)m * N + n] = (__bf16)v;
      }
}

// ------------------------------- RoPE (in place, bf16) ---------------------
__global__ __launch_bounds__(256)
void rope_kernel(__bf16* __restrict__ data, int rows, int heads, int ld) {
  const int idx = blockIdx.x * 256 + threadIdx.x;
  const int total = rows * heads * (HEAD_DIM / 2);
  if (idx >= total) return;
  const int i   = idx & 31;                  // rotary pair index 0..31
  const int h   = (idx >> 5) % heads;
  const int row = idx / (32 * heads);
  const int t   = row & (T_SEQ - 1);         // position within sequence
  const float inv = __powf(10000.0f, -(float)i / 32.0f);
  const float ang = (float)t * inv;
  float sn, cs;
  __sincosf(ang, &sn, &cs);
  __bf16* p = data + (size_t)row * ld + h * HEAD_DIM;
  const float x1 = (float)p[i];
  const float x2 = (float)p[i + 32];
  p[i]      = (__bf16)(x1 * cs - x2 * sn);
  p[i + 32] = (__bf16)(x2 * cs + x1 * sn);
}

// ------------------------------- Flash attention ---------------------------
// grid = (BATCH*N_HEADS, T/64); block = 128 (4 waves), each wave owns 16 q rows.
__global__ __launch_bounds__(128)
void flash_attn_wmma(const __bf16* __restrict__ qb, const __bf16* __restrict__ kb,
                     const __bf16* __restrict__ vb, __bf16* __restrict__ ctxb) {
  const int h      = blockIdx.x & (N_HEADS - 1);
  const int bb     = blockIdx.x >> 5;
  const int qblock = blockIdx.y;             // 64-row q block
  const int kvh    = h >> 2;                 // G = 4

  __shared__ __align__(16) __bf16 Kt[64 * 64];
  __shared__ __align__(16) __bf16 Vt[64 * 64];
  __shared__ __align__(16) __bf16 Pt[4][16 * 64];

  const int tid  = threadIdx.x;
  const int wave = tid >> 5, lane = tid & 31;
  const int half = lane >> 4, l16 = lane & 15;
  const float scale = 0.125f;                // 1/sqrt(64)

  // Q fragments for this wave's 16 rows (A layout), 2 chunks over HEAD_DIM
  v16bf aQ[2];
  const int qrow = bb * T_SEQ + qblock * 64 + wave * 16 + l16;
  #pragma unroll
  for (int kc = 0; kc < 2; ++kc)
    #pragma unroll
    for (int e = 0; e < 16; ++e) {
      const int kk = kc * 32 + half * 16 + e;
      aQ[kc][e] = qb[(size_t)qrow * D_MODEL + h * HEAD_DIM + kk];
    }

  v8f acc[4] = {};
  float mrun[8], lrun[8];
  #pragma unroll
  for (int r = 0; r < 8; ++r) { mrun[r] = -1e30f; lrun[r] = 0.0f; }

  for (int kbk = 0; kbk <= qblock; ++kbk) {
    // K/V tiles 64x64 bf16 = 512 16B chunks each; 128 threads -> 4 chunks each
    #pragma unroll
    for (int c = 0; c < 4; ++c) {
      const int i = tid + c * 128;             // chunk id 0..511
      const int r = i >> 3, col = (i & 7) * 8;
      const size_t g =
          (size_t)(bb * T_SEQ + kbk * 64 + r) * KV_D + kvh * HEAD_DIM + col;
      async_copy16(&kb[g], &Kt[r * 64 + col]);
      async_copy16(&vb[g], &Vt[r * 64 + col]);
    }
    if (kbk < qblock) {  // prefetch next K tile (global_prefetch_b8)
      const size_t g = (size_t)(bb * T_SEQ + (kbk + 1) * 64 + (tid >> 1)) * KV_D +
                       kvh * HEAD_DIM + (tid & 1) * 32;
      __builtin_prefetch(&kb[g], 0, 1);
    }
    wait_async0();
    __syncthreads();

    // S = Q @ K^T  (16 x 64 per wave)
    v8f s[4] = {};
    #pragma unroll
    for (int jc = 0; jc < 4; ++jc) {
      #pragma unroll
      for (int kc = 0; kc < 2; ++kc) {
        v16bf bK;
        const int j = jc * 16 + l16;
        #pragma unroll
        for (int e = 0; e < 16; ++e) {
          const int kd = kc * 32 + half * 16 + e;
          bK[e] = Kt[j * 64 + kd];           // K^T[kd][j] = K[j][kd]
        }
        s[jc] = __builtin_amdgcn_wmma_f32_16x16x32_bf16(
            false, aQ[kc], false, bK, (short)0, s[jc], false, false);
      }
    }

    // online softmax per row (row = r + 8*half, spread across 16 lanes)
    #pragma unroll
    for (int r = 0; r < 8; ++r) {
      const int qi = qblock * 64 + wave * 16 + r + 8 * half;
      float mx = mrun[r];
      #pragma unroll
      for (int jc = 0; jc < 4; ++jc) {
        const int kvj = kbk * 64 + jc * 16 + l16;
        const float sv = (kvj <= qi) ? s[jc][r] * scale : -1e30f;
        s[jc][r] = sv;
        mx = fmaxf(mx, sv);
      }
      #pragma unroll
      for (int off = 1; off < 16; off <<= 1)
        mx = fmaxf(mx, __shfl_xor(mx, off, 32));
      const float alpha = __expf(mrun[r] - mx);
      float ps = 0.0f;
      #pragma unroll
      for (int jc = 0; jc < 4; ++jc) {
        const float p = __expf(s[jc][r] - mx);
        s[jc][r] = p;
        ps += p;
      }
      #pragma unroll
      for (int off = 1; off < 16; off <<= 1)
        ps += __shfl_xor(ps, off, 32);
      lrun[r] = lrun[r] * alpha + ps;
      mrun[r] = mx;
      #pragma unroll
      for (int nf = 0; nf < 4; ++nf) acc[nf][r] *= alpha;
    }

    // C-layout -> A-layout via LDS for P
    #pragma unroll
    for (int jc = 0; jc < 4; ++jc)
      #pragma unroll
      for (int r = 0; r < 8; ++r)
        Pt[wave][(r + 8 * half) * 64 + jc * 16 + l16] = (__bf16)s[jc][r];
    __syncthreads();

    // ctx += P @ V
    #pragma unroll
    for (int kc2 = 0; kc2 < 2; ++kc2) {
      v16bf aP;
      #pragma unroll
      for (int e = 0; e < 16; ++e) {
        const int kk = kc2 * 32 + half * 16 + e;
        aP[e] = Pt[wave][l16 * 64 + kk];
      }
      #pragma unroll
      for (int nf = 0; nf < 4; ++nf) {
        v16bf bV;
        #pragma unroll
        for (int e = 0; e < 16; ++e) {
          const int kd = kc2 * 32 + half * 16 + e;
          bV[e] = Vt[kd * 64 + nf * 16 + l16];
        }
        acc[nf] = __builtin_amdgcn_wmma_f32_16x16x32_bf16(
            false, aP, false, bV, (short)0, acc[nf], false, false);
      }
    }
    __syncthreads();   // protect Kt/Vt for next iteration
  }

  // normalize + write ctx (bf16), layout [b*T, D] with head columns
  #pragma unroll
  for (int nf = 0; nf < 4; ++nf)
    #pragma unroll
    for (int r = 0; r < 8; ++r) {
      const int rowT = qblock * 64 + wave * 16 + r + 8 * half;
      const float v = acc[nf][r] / lrun[r];
      ctxb[(size_t)(bb * T_SEQ + rowT) * D_MODEL + h * HEAD_DIM + nf * 16 + l16] =
          (__bf16)v;
    }
}

// ------------------------------- launch ------------------------------------
extern "C" void kernel_launch(void* const* d_in, const int* in_sizes, int n_in,
                              void* d_out, int out_size, void* d_ws, size_t ws_size,
                              hipStream_t stream) {
  const float* x    = (const float*)d_in[0];
  const float* Wq   = (const float*)d_in[1];
  const float* Wk   = (const float*)d_in[2];
  const float* Wv   = (const float*)d_in[3];
  const float* Wo   = (const float*)d_in[4];
  const float* bout = (const float*)d_in[5];
  float* out = (float*)d_out;
  (void)in_sizes; (void)n_in; (void)out_size;

  const size_t NX  = (size_t)M_ROWS * D_MODEL;       // 8,388,608
  const size_t NWQ = (size_t)D_MODEL * D_MODEL;
  const size_t NWK = (size_t)D_MODEL * KV_D;
  const size_t NQ  = (size_t)M_ROWS * D_MODEL;
  const size_t NK  = (size_t)M_ROWS * KV_D;

  char* ws = (char*)d_ws;
  size_t off = 0;
  auto bump = [&](size_t bytes) { size_t r = off; off += (bytes + 255) & ~(size_t)255; return r; };
  __bf16* xb   = (__bf16*)(ws + bump(NX  * 2));
  __bf16* Wqb  = (__bf16*)(ws + bump(NWQ * 2));
  __bf16* Wkb  = (__bf16*)(ws + bump(NWK * 2));
  __bf16* Wvb  = (__bf16*)(ws + bump(NWK * 2));
  __bf16* Wob  = (__bf16*)(ws + bump(NWQ * 2));
  __bf16* qb   = (__bf16*)(ws + bump(NQ  * 2));
  __bf16* kbuf = (__bf16*)(ws + bump(NK  * 2));
  __bf16* vbuf = (__bf16*)(ws + bump(NK  * 2));
  __bf16* ctxb = (__bf16*)(ws + bump(NQ  * 2));
  if (off > ws_size) return;

  // 1) casts
  cast_f32_bf16<<<(int)((NX  + 255) / 256), 256, 0, stream>>>(x,  xb,  (int)NX);
  cast_f32_bf16<<<(int)((NWQ + 255) / 256), 256, 0, stream>>>(Wq, Wqb, (int)NWQ);
  cast_f32_bf16<<<(int)((NWK + 255) / 256), 256, 0, stream>>>(Wk, Wkb, (int)NWK);
  cast_f32_bf16<<<(int)((NWK + 255) / 256), 256, 0, stream>>>(Wv, Wvb, (int)NWK);
  cast_f32_bf16<<<(int)((NWQ + 255) / 256), 256, 0, stream>>>(Wo, Wob, (int)NWQ);

  // 2) QKV projections (WMMA GEMM, async-LDS double buffered)
  gemm_bf16_wmma<false><<<dim3(D_MODEL / 128, M_ROWS / 128), 256, 0, stream>>>(
      xb, Wqb, qb, nullptr, nullptr, M_ROWS, D_MODEL, D_MODEL);
  gemm_bf16_wmma<false><<<dim3(KV_D / 128, M_ROWS / 128), 256, 0, stream>>>(
      xb, Wkb, kbuf, nullptr, nullptr, M_ROWS, KV_D, D_MODEL);
  gemm_bf16_wmma<false><<<dim3(KV_D / 128, M_ROWS / 128), 256, 0, stream>>>(
      xb, Wvb, vbuf, nullptr, nullptr, M_ROWS, KV_D, D_MODEL);

  // 3) RoPE on q and k
  {
    const int nq = M_ROWS * N_HEADS * 32;
    const int nk = M_ROWS * KV_HEADS * 32;
    rope_kernel<<<(nq + 255) / 256, 256, 0, stream>>>(qb, M_ROWS, N_HEADS, D_MODEL);
    rope_kernel<<<(nk + 255) / 256, 256, 0, stream>>>(kbuf, M_ROWS, KV_HEADS, KV_D);
  }

  // 4) flash attention (WMMA, online softmax, causal, async K/V staging)
  flash_attn_wmma<<<dim3(BATCH * N_HEADS, T_SEQ / 64), 128, 0, stream>>>(
      qb, kbuf, vbuf, ctxb);

  // 5) output projection + bias (f32 out)
  gemm_bf16_wmma<true><<<dim3(D_MODEL / 128, M_ROWS / 128), 256, 0, stream>>>(
      ctxb, Wob, nullptr, out, bout, M_ROWS, D_MODEL, D_MODEL);
}
